// GroupedQueryAttention_10204842295766
// MI455X (gfx1250) — matrix operand order repaired and checked by reading, not compiled, not run on hardware
//
#include <hip/hip_runtime.h>
#include <hip/hip_bf16.h>
#include <stddef.h>

// ---------------------------------------------------------------------------
// GQA (B=2, T=1024, D=2048, H=32, G=8, HD=64) on gfx1250 via bf16 WMMA.
// Pipeline: 3x projection GEMMs -> flash attention -> output GEMM.
// GEMM: 128x128 tile / 8 waves / K-chunk 64 (16 WMMA per wave per sync).
// Attention: 64-key staging blocks (16 WMMA per wave per sync).
// ---------------------------------------------------------------------------

typedef __attribute__((ext_vector_type(16))) __bf16 v16bf;
typedef __attribute__((ext_vector_type(8)))  __bf16 v8bf;
typedef __attribute__((ext_vector_type(8)))  float  v8f;

__device__ __forceinline__ __bf16 f2bf(float f) {
    union { float f; unsigned u; } a; a.f = f;
    unsigned r = a.u + 0x7FFFu + ((a.u >> 16) & 1u);   // round-to-nearest-even
    unsigned short hs = (unsigned short)(r >> 16);
    union { unsigned short s; __bf16 h; } b; b.s = hs;
    return b.h;
}

__device__ __forceinline__ v8f wmma_bf16(v16bf a, v16bf b, v8f c) {
    // D = A(16x32 bf16) x B(32x16 bf16) + C(16x16 f32)
    return __builtin_amdgcn_wmma_f32_16x16x32_bf16(
        false, a, false, b, (short)0, c, false, false);
}

// A-fragment (16x32, 16-bit): lane M = L%16; half = L/16 selects K sub-block
// [half*16, half*16+16) of a 32-wide row.
__device__ __forceinline__ v16bf load_a_frag(const __bf16* row, int half) {
    return *(const v16bf*)(row + half * 16);
}

// ---------------------------------------------------------------------------
// GEMM: C[M,N] = A[M,K] * B[K,N], bf16 WMMA, fp32 accumulate.
// 256 threads (8 waves), 128x128 C tile, K staged in chunks of 64.
// Wave w computes rows [w*16, w*16+16) x all 128 columns (8 N-tiles).
// ---------------------------------------------------------------------------
template <bool A_BF16, bool OUT_BF16>
__global__ __launch_bounds__(256)
void gemm_bf16_128x128(const void* __restrict__ Ap, const float* __restrict__ Bw,
                       void* __restrict__ Cp, int M, int N, int K) {
    __shared__ __align__(128) __bf16 Alds[128][64];   // row-major (A-frag layout)
    __shared__ __align__(128) __bf16 BldsT[128][64];  // [n][k] (B-frag contiguous)
    const int tid = threadIdx.x;
    const int w = tid >> 5, L = tid & 31;
    const int half = L >> 4, lm = L & 15;
    const int m0 = blockIdx.y * 128, n0 = blockIdx.x * 128;
    const float*  Af = (const float*)Ap;
    const __bf16* Ab = (const __bf16*)Ap;
    v8f acc[8] = {};

    for (int kt = 0; kt < K; kt += 64) {
        __syncthreads();
#pragma unroll
        for (int i = 0; i < 32; ++i) {           // stage 128x64 A and 64x128 B
            int e  = i * 256 + tid;
            int ar = e >> 6, ac = e & 63;
            Alds[ar][ac] = A_BF16 ? Ab[(size_t)(m0 + ar) * K + kt + ac]
                                  : f2bf(Af[(size_t)(m0 + ar) * K + kt + ac]);
            int kk = e >> 7, nn = e & 127;
            BldsT[nn][kk] = f2bf(Bw[(size_t)(kt + kk) * N + n0 + nn]);
        }
        if (kt + 64 < K) {                       // prefetch next K-chunk into L2
            int e  = tid;
            int ar = e >> 1, ac = (e & 1) * 32;
            if (A_BF16) __builtin_prefetch(Ab + (size_t)(m0 + ar) * K + kt + 64 + ac, 0, 2);
            else        __builtin_prefetch(Af + (size_t)(m0 + ar) * K + kt + 64 + ac, 0, 2);
            int kk = e >> 2, nn = (e & 3) * 32;
            __builtin_prefetch(Bw + (size_t)(kt + 64 + kk) * N + n0 + nn, 0, 2);
        }
        __syncthreads();
        v16bf af[2];
#pragma unroll
        for (int kc = 0; kc < 2; ++kc)
            af[kc] = load_a_frag(&Alds[w * 16 + lm][kc * 32], half);
#pragma unroll
        for (int t = 0; t < 8; ++t)
#pragma unroll
            for (int kc = 0; kc < 2; ++kc) {
                v16bf bf = *(const v16bf*)(&BldsT[t * 16 + lm][kc * 32 + half * 16]);
                acc[t] = wmma_bf16(af[kc], bf, acc[t]);
            }
    }
#pragma unroll
    for (int t = 0; t < 8; ++t)
#pragma unroll
        for (int r = 0; r < 8; ++r) {
            size_t idx = (size_t)(m0 + w * 16 + r + 8 * half) * N + n0 + t * 16 + lm;
            float vv = acc[t][r];
            if (OUT_BF16) ((__bf16*)Cp)[idx] = f2bf(vv);
            else          ((float*)Cp)[idx]  = vv;
        }
}

// ---------------------------------------------------------------------------
// Flash attention, causal, grouped KV.  One wave = one 16-row Q tile; 4 waves
// (64 rows of one head) share 64-key K/V blocks in LDS.  V stored transposed.
// ---------------------------------------------------------------------------
__global__ __launch_bounds__(128)
void gqa_flash(const void* __restrict__ qv, const void* __restrict__ kv,
               const void* __restrict__ vv, void* __restrict__ ov) {
    constexpr int T = 1024, D = 2048, KVW = 512, HD = 64;
    const __bf16* q = (const __bf16*)qv;
    const __bf16* k = (const __bf16*)kv;
    const __bf16* v = (const __bf16*)vv;
    __bf16*       o = (__bf16*)ov;

    __shared__ __align__(128) __bf16 Klds[64][64];    // [key][d]
    __shared__ __align__(128) __bf16 VldsT[64][64];   // [d][key]
    __shared__ __align__(128) __bf16 Plds[4][16][64]; // per-wave P scratch

    const int tid = threadIdx.x;
    const int w = tid >> 5, L = tid & 31;
    const int half = L >> 4, lm = L & 15;
    const int bx   = blockIdx.x;
    const int qblk = bx & 15;
    const int h    = (bx >> 4) & 31;
    const int b    = bx >> 9;
    const int g    = h >> 2;          // kv group (GS = 4)
    const int qb   = qblk * 64;

    // Q fragments (16 rows x 64 d), two K=32 chunks
    const size_t qrow0 = ((size_t)(b * T) + qb + w * 16 + lm) * D + (size_t)h * HD;
    v16bf aq[2];
#pragma unroll
    for (int c = 0; c < 2; ++c) aq[c] = load_a_frag(q + qrow0 + c * 32, half);

    float mrow[8], lrow[8];
#pragma unroll
    for (int r = 0; r < 8; ++r) { mrow[r] = -1e30f; lrow[r] = 0.0f; }
    v8f acc[4] = {};

    const int myq = qb + w * 16;
    for (int jb = 0; jb <= qblk; ++jb) {         // 64-key blocks over [0, qb+64)
        const int kbase = jb * 64;
        __syncthreads();
#pragma unroll
        for (int i = 0; i < 32; ++i) {           // stage 64x64 K and V
            int e  = i * 128 + tid;
            int kk = e >> 6, dd = e & 63;
            size_t gi = ((size_t)(b * T) + kbase + kk) * KVW + (size_t)g * HD + dd;
            Klds[kk][dd]  = k[gi];
            VldsT[dd][kk] = v[gi];
        }
        if (jb < qblk) {                         // prefetch next K/V block
            int kk = tid >> 1, dd = (tid & 1) * 32;
            size_t gi = ((size_t)(b * T) + kbase + 64 + kk) * KVW + (size_t)g * HD + dd;
            __builtin_prefetch(k + gi, 0, 2);
            __builtin_prefetch(v + gi, 0, 2);
        }
        __syncthreads();

        // S = Q * K^T  (four 16x16 tiles covering 64 keys)
        v8f s[4];
#pragma unroll
        for (int st = 0; st < 4; ++st) {
            v8f z = {};
            v16bf bk0 = *(const v16bf*)(&Klds[st * 16 + lm][half * 16]);
            v16bf bk1 = *(const v16bf*)(&Klds[st * 16 + lm][32 + half * 16]);
            z = wmma_bf16(aq[0], bk0, z);
            z = wmma_bf16(aq[1], bk1, z);
            s[st] = z;
        }

        const float scale = 0.125f;              // 1/sqrt(64)
#pragma unroll
        for (int r = 0; r < 8; ++r) {
            const int qrow = myq + r + 8 * half;
            float x[4], mx = -1e30f;
#pragma unroll
            for (int st = 0; st < 4; ++st) {
                x[st] = s[st][r] * scale;
                if (kbase + st * 16 + lm > qrow) x[st] = -1e30f;  // causal mask
                mx = fmaxf(mx, x[st]);
            }
#pragma unroll
            for (int d = 1; d < 16; d <<= 1) mx = fmaxf(mx, __shfl_xor(mx, d, 32));
            float mnew = fmaxf(mrow[r], mx);
            float p[4], rs = 0.0f;
#pragma unroll
            for (int st = 0; st < 4; ++st) { p[st] = __expf(x[st] - mnew); rs += p[st]; }
#pragma unroll
            for (int d = 1; d < 16; d <<= 1) rs += __shfl_xor(rs, d, 32);
            float alpha = __expf(mrow[r] - mnew);
            lrow[r] = lrow[r] * alpha + rs;
            mrow[r] = mnew;
#pragma unroll
            for (int t = 0; t < 4; ++t) acc[t][r] = acc[t][r] * alpha;
#pragma unroll
            for (int st = 0; st < 4; ++st)       // C-layout -> row major bf16
                Plds[w][r + 8 * half][st * 16 + lm] = f2bf(p[st]);
        }

        // O += P(16x64) * V(64x64)
        v16bf ap[2];
#pragma unroll
        for (int kc = 0; kc < 2; ++kc)
            ap[kc] = load_a_frag(&Plds[w][lm][kc * 32], half);
#pragma unroll
        for (int t = 0; t < 4; ++t)
#pragma unroll
            for (int kc = 0; kc < 2; ++kc) {
                v16bf bv = *(const v16bf*)(&VldsT[t * 16 + lm][kc * 32 + half * 16]);
                acc[t] = wmma_bf16(ap[kc], bv, acc[t]);
            }
    }

    // normalize and write out (bf16, layout [b, t, h*64 + d])
#pragma unroll
    for (int r = 0; r < 8; ++r) {
        float inv = 1.0f / lrow[r];
        size_t orow = ((size_t)(b * T) + qb + w * 16 + r + 8 * half) * D + (size_t)h * HD;
#pragma unroll
        for (int t = 0; t < 4; ++t)
            o[orow + t * 16 + lm] = f2bf(acc[t][r] * inv);
    }
}

// ---------------------------------------------------------------------------
extern "C" void kernel_launch(void* const* d_in, const int* in_sizes, int n_in,
                              void* d_out, int out_size, void* d_ws, size_t ws_size,
                              hipStream_t stream) {
    (void)in_sizes; (void)n_in; (void)out_size; (void)ws_size;
    const float* x  = (const float*)d_in[0];
    const float* Wq = (const float*)d_in[1];
    const float* Wk = (const float*)d_in[2];
    const float* Wv = (const float*)d_in[3];
    const float* Wo = (const float*)d_in[4];
    float* out = (float*)d_out;

    const int Mr = 2048;   // B*T
    const int D  = 2048;
    const int KV = 512;    // G*HD

    // ws carve (bytes): q 8MB | k 2MB | v 2MB | attn 8MB  (= 20 MB total)
    char* w0 = (char*)d_ws;
    void* qb = (void*)(w0);
    void* kb = (void*)(w0 + (size_t)Mr * D * 2);
    void* vb = (void*)(w0 + (size_t)Mr * D * 2 + (size_t)Mr * KV * 2);
    void* ab = (void*)(w0 + (size_t)Mr * D * 2 + (size_t)Mr * KV * 4);

    dim3 blk(256);
    gemm_bf16_128x128<false, true><<<dim3(D / 128,  Mr / 128), blk, 0, stream>>>(x,  Wq, qb, Mr, D,  D);
    gemm_bf16_128x128<false, true><<<dim3(KV / 128, Mr / 128), blk, 0, stream>>>(x,  Wk, kb, Mr, KV, D);
    gemm_bf16_128x128<false, true><<<dim3(KV / 128, Mr / 128), blk, 0, stream>>>(x,  Wv, vb, Mr, KV, D);
    gqa_flash<<<dim3(1024), dim3(128), 0, stream>>>(qb, kb, vb, ab);
    gemm_bf16_128x128<true, false><<<dim3(D / 128,  Mr / 128), blk, 0, stream>>>(ab, Wo, out, Mr, D, D);
}
